// PowerFlowSAGE_61177514164375
// MI455X (gfx1250) — matrix operand order repaired and checked by reading, not compiled, not run on hardware
//
#include <hip/hip_runtime.h>

// ---------------------------------------------------------------------------
// PowerFlow GraphSAGE for MI455X (gfx1250, wave32, WMMA bf16->f32)
//
// Activations live in global memory as bf16 (the WMMA path truncates to bf16
// anyway), so the edge-gather working set (h: 76.8MB) is L2-resident on the
// 192MB L2 and per-layer gather traffic is halved. Mean-aggregation stays in
// f32 (exact f32 atomics); the 1/deg scale + f32->bf16 pack is fused into the
// GEMM's LDS staging.
// ---------------------------------------------------------------------------
#define N_NODES 100000
#define N_EDGES 300000
#define IN_DIM  10
#define HID     384
#define OUT_DIM 3
#define T_LAYERS 5
#define BN_EPS  1e-5f

#define TILES  24      // 384 / 16 output-column tiles
#define KSTEPS 12      // 384 / 32 K-steps
#define ROWPAD 392     // u16 per LDS row: stride 784B = 49*16B -> ds_load_b128 conflict-free

typedef unsigned int  u32;
typedef unsigned short u16;
typedef __attribute__((ext_vector_type(16))) __bf16 v16bf;
typedef __attribute__((ext_vector_type(8)))  float  v8f;
typedef __attribute__((ext_vector_type(4)))  u32    u32x4;

union Frag { v16bf v; u32x4 q[2]; };

// f32 -> bf16 (round-to-nearest-even), kept as raw u16 storage
__device__ __forceinline__ u16 f2bf(float f) {
  u32 u = __float_as_uint(f);
  u32 r = ((u >> 16) & 1u) + 0x7FFFu;
  return (u16)((u + r) >> 16);
}
__device__ __forceinline__ float bf2f(u16 h) {
  return __uint_as_float((u32)h << 16);
}
__device__ __forceinline__ u32 pack2(float a, float b) {
  return (u32)f2bf(a) | ((u32)f2bf(b) << 16);
}

// ---------------------------------------------------------------------------
// utility kernels
// ---------------------------------------------------------------------------
__global__ void zero_kernel(float* __restrict__ p, long n) {
  long i = (long)blockIdx.x * blockDim.x + threadIdx.x;
  long stride = (long)gridDim.x * blockDim.x;
  for (; i < n; i += stride) p[i] = 0.0f;
}

__global__ void count_deg(const int* __restrict__ dst, float* __restrict__ deg) {
  int i = blockIdx.x * 256 + threadIdx.x;
  if (i < N_EDGES) atomicAdd(&deg[dst[i]], 1.0f);
}

__global__ void scatter10(const int* __restrict__ src, const int* __restrict__ dst,
                          const float* __restrict__ x, float* __restrict__ agg) {
  int i = blockIdx.x * 256 + threadIdx.x;
  if (i < N_EDGES * IN_DIM) {
    int e = i / IN_DIM;
    int f = i - e * IN_DIM;
    atomicAdd(&agg[(size_t)dst[e] * IN_DIM + f], x[(size_t)src[e] * IN_DIM + f]);
  }
}

// one block per edge: gather a bf16 row (L2-resident), f32-atomic into agg.
// 64 threads x gridDim.y=3 chunks -> each thread one bf16 pair (u32 load).
__global__ __launch_bounds__(64)
void scatter384(const int* __restrict__ src, const int* __restrict__ dst,
                const u16* __restrict__ h, float* __restrict__ agg) {
  const int e = blockIdx.x;
  const int p = blockIdx.y * 64 + threadIdx.x;   // pair index 0..191
  const int s = src[e];
  const int d = dst[e];
  const u32 pair = ((const u32*)(h + (size_t)s * HID))[p];
  float* ap = &agg[(size_t)d * HID + 2 * p];
  atomicAdd(ap,     bf2f((u16)(pair & 0xFFFFu)));
  atomicAdd(ap + 1, bf2f((u16)(pair >> 16)));
}

// ---------------------------------------------------------------------------
// Pack W[384x384] (out-major) into WMMA B-fragment order (bf16):
//   layout[tile_c][kstep][lane][16 elems], 32B contiguous per lane
//   B(k, n): n = tile_c*16 + (lane&15); k = kstep*32 + ((lane&16)?8:0) + (i<8 ? i : i+8)
// ---------------------------------------------------------------------------
__global__ __launch_bounds__(256)
void pack_weight(const float* __restrict__ W, u16* __restrict__ out) {
  int p    = blockIdx.x * 256 + threadIdx.x;    // 24*12*32*16 = 147456 total
  int i    = p & 15;
  int lane = (p >> 4) & 31;
  int ks   = (p >> 9) % KSTEPS;
  int tile = p / (512 * KSTEPS);
  int n = tile * 16 + (lane & 15);
  int k = ks * 32 + ((lane & 16) ? 8 : 0) + (i < 8 ? i : i + 8);
  out[p] = f2bf(W[(size_t)n * HID + k]);
}

// ---------------------------------------------------------------------------
// Fused GEMM + (bias, [L2-norm], BN, activation).
//   SAGE=true : out = ELU(BN(l2norm(agg/deg @ Wl^T + b + h @ Wr^T)))   (agg f32, h bf16)
//   SAGE=false: out = ReLU(BN(h @ Wp^T + b))                           (h bf16)
// Block: 16 rows x 384 cols, 256 threads = 8 waves, 3 16x16 tiles per wave.
// A fragments via ds_load_b128 from padded bf16 LDS tiles, B fragments via
// global_load_b128 from pre-packed weights; f32 accumulate; bf16 output.
// ---------------------------------------------------------------------------
template<bool SAGE>
__global__ __launch_bounds__(256)
void gemm_bn_act(const float* __restrict__ aggf,  // SAGE only: f32 neighbor sums
                 const u16*   __restrict__ hb,    // bf16 activations (x in SAGE, input in MLP)
                 const float* __restrict__ deg,   // SAGE only
                 const u16*   __restrict__ pB0,   // packed Wl / Wp
                 const u16*   __restrict__ pB1,   // packed Wr (SAGE only)
                 const float* __restrict__ bias,
                 const float* __restrict__ bng, const float* __restrict__ bnb,
                 const float* __restrict__ bnm, const float* __restrict__ bnv,
                 u16* __restrict__ out) {
  __shared__ __align__(16) u16 tA[16][ROWPAD];
  __shared__ __align__(16) u16 tX[SAGE ? 16 : 1][ROWPAD];
  __shared__ float rowsq[16];
  __shared__ float invd[16];

  const int tid   = threadIdx.x;
  const int node0 = blockIdx.x << 4;

  if (tid < 16) {
    rowsq[tid] = 0.0f;
    if (SAGE) invd[tid] = 1.0f / fmaxf(deg[node0 + tid], 1.0f);
  }
  __syncthreads();

  // stage activation tiles into LDS in 16B chunks (48 chunks per 384-col row)
  for (int w = tid; w < 16 * 48; w += 256) {
    const int r = w / 48;
    const int c = w - r * 48;                       // 16B chunk = 8 bf16 features
    u32* ldsA = (u32*)&tA[0][0] + r * (ROWPAD / 2) + c * 4;
    if (SAGE) {
      const float iv = invd[r];
      const u32x4* af = (const u32x4*)(aggf + (size_t)(node0 + r) * HID + c * 8);
      u32x4 f0 = af[0];
      u32x4 f1 = af[1];
      u32x4 v;
      v.x = pack2(__uint_as_float(f0.x) * iv, __uint_as_float(f0.y) * iv);
      v.y = pack2(__uint_as_float(f0.z) * iv, __uint_as_float(f0.w) * iv);
      v.z = pack2(__uint_as_float(f1.x) * iv, __uint_as_float(f1.y) * iv);
      v.w = pack2(__uint_as_float(f1.z) * iv, __uint_as_float(f1.w) * iv);
      *(u32x4*)ldsA = v;
      // x-tile: pure bf16 b128 copy
      u32* ldsX = (u32*)&tX[0][0] + r * (ROWPAD / 2) + c * 4;
      *(u32x4*)ldsX = ((const u32x4*)(hb + (size_t)(node0 + r) * HID))[c];
    } else {
      *(u32x4*)ldsA = ((const u32x4*)(hb + (size_t)(node0 + r) * HID))[c];
    }
  }
  __syncthreads();

  const int lane   = tid & 31;
  const int wv     = tid >> 5;
  const int nsub   = lane & 15;            // A: row m; B/C/D: col n
  const int khalf  = (lane & 16) ? 8 : 0;  // A layout: lanes 16-31 start at K=8
  const int rowoff = (lane & 16) ? 8 : 0;  // C/D layout: lanes 16-31 hold M=8..15

  float res[3][8];

  for (int j = 0; j < 3; ++j) {
    const int c = wv + (j << 3);           // output tile column index (0..23)
    v8f acc = {};
    const u16* bbase0 = pB0 + (size_t)c * KSTEPS * 512 + (size_t)lane * 16;
    for (int ks = 0; ks < KSTEPS; ++ks) {
      Frag a, b;
      const u16* ap = &tA[nsub][ks * 32 + khalf];
      a.q[0] = *(const u32x4*)(ap);        // K = khalf + 0..7
      a.q[1] = *(const u32x4*)(ap + 16);   // K = khalf + 16..23
      const u16* bp = bbase0 + (size_t)ks * 512;
      b.q[0] = *(const u32x4*)(bp);
      b.q[1] = *(const u32x4*)(bp + 8);
      acc = __builtin_amdgcn_wmma_f32_16x16x32_bf16(false, a.v, false, b.v,
                                                    (short)0, acc, false, false);
    }
    if (SAGE) {
      const u16* bbase1 = pB1 + (size_t)c * KSTEPS * 512 + (size_t)lane * 16;
      for (int ks = 0; ks < KSTEPS; ++ks) {
        Frag a, b;
        const u16* ap = &tX[nsub][ks * 32 + khalf];
        a.q[0] = *(const u32x4*)(ap);
        a.q[1] = *(const u32x4*)(ap + 16);
        const u16* bp = bbase1 + (size_t)ks * 512;
        b.q[0] = *(const u32x4*)(bp);
        b.q[1] = *(const u32x4*)(bp + 8);
        acc = __builtin_amdgcn_wmma_f32_16x16x32_bf16(false, a.v, false, b.v,
                                                      (short)0, acc, false, false);
      }
    }
    const float bs = bias[c * 16 + nsub];
    for (int r = 0; r < 8; ++r) res[j][r] = acc[r] + bs;
  }

  if (SAGE) {
    // per-row sum of squares: reduce across the 16 column-lanes, then LDS atomics
    float s[8];
    for (int r = 0; r < 8; ++r) {
      float v = res[0][r] * res[0][r] + res[1][r] * res[1][r] + res[2][r] * res[2][r];
      v += __shfl_xor(v, 1);
      v += __shfl_xor(v, 2);
      v += __shfl_xor(v, 4);
      v += __shfl_xor(v, 8);
      s[r] = v;
    }
    if (nsub == 0)
      for (int r = 0; r < 8; ++r) atomicAdd(&rowsq[rowoff + r], s[r]);
    __syncthreads();
  }

  float scale[8];
  if (SAGE)
    for (int r = 0; r < 8; ++r)
      scale[r] = 1.0f / fmaxf(sqrtf(rowsq[rowoff + r]), 1e-12f);

  for (int j = 0; j < 3; ++j) {
    const int col = (wv + (j << 3)) * 16 + nsub;
    const float g  = bng[col];
    const float bb = bnb[col];
    const float mu = bnm[col];
    const float is = rsqrtf(bnv[col] + BN_EPS);
    for (int r = 0; r < 8; ++r) {
      float y = res[j][r];
      if (SAGE) y *= scale[r];
      y = (y - mu) * is * g + bb;
      if (SAGE) y = (y > 0.0f) ? y : (__expf(y) - 1.0f);   // ELU
      else      y = fmaxf(y, 0.0f);                        // ReLU
      out[(size_t)(node0 + rowoff + r) * HID + col] = f2bf(y);
    }
  }
}

// ---------------------------------------------------------------------------
// Layer 0: K=10 is too skinny for WMMA tiles -> VALU, one block per node.
// ---------------------------------------------------------------------------
__global__ __launch_bounds__(128)
void layer0_kernel(const float* __restrict__ x, const float* __restrict__ agg10,
                   const float* __restrict__ deg,
                   const float* __restrict__ Wl0, const float* __restrict__ bl0,
                   const float* __restrict__ Wr0,
                   const float* __restrict__ bng, const float* __restrict__ bnb,
                   const float* __restrict__ bnm, const float* __restrict__ bnv,
                   u16* __restrict__ out) {
  __shared__ float xr[IN_DIM], ar[IN_DIM];
  __shared__ float ssq;
  const int node = blockIdx.x;
  const int tid  = threadIdx.x;
  if (tid == 0) ssq = 0.0f;
  if (tid < IN_DIM) {
    xr[tid] = x[(size_t)node * IN_DIM + tid];
    ar[tid] = agg10[(size_t)node * IN_DIM + tid] / fmaxf(deg[node], 1.0f);
  }
  __syncthreads();
  float v[3];
  float s = 0.0f;
  for (int j = 0; j < 3; ++j) {
    const int o = tid + (j << 7);
    float acc = bl0[o];
    for (int k = 0; k < IN_DIM; ++k)
      acc += ar[k] * Wl0[o * IN_DIM + k] + xr[k] * Wr0[o * IN_DIM + k];
    v[j] = acc;
    s += acc * acc;
  }
  s += __shfl_xor(s, 1);  s += __shfl_xor(s, 2);  s += __shfl_xor(s, 4);
  s += __shfl_xor(s, 8);  s += __shfl_xor(s, 16);
  if ((tid & 31) == 0) atomicAdd(&ssq, s);
  __syncthreads();
  const float scale = 1.0f / fmaxf(sqrtf(ssq), 1e-12f);
  for (int j = 0; j < 3; ++j) {
    const int o = tid + (j << 7);
    float y = v[j] * scale;
    y = (y - bnm[o]) * rsqrtf(bnv[o] + BN_EPS) * bng[o] + bnb[o];
    y = (y > 0.0f) ? y : (__expf(y) - 1.0f);
    out[(size_t)node * HID + o] = f2bf(y);
  }
}

// ---------------------------------------------------------------------------
// Final 384 -> 3 projection: one block per node, wave-reduced dots (h is bf16).
// ---------------------------------------------------------------------------
__global__ __launch_bounds__(128)
void proj3(const u16* __restrict__ h, const float* __restrict__ Wp2,
           const float* __restrict__ bp2, float* __restrict__ out) {
  __shared__ float acc3[OUT_DIM];
  const int node = blockIdx.x;
  const int tid  = threadIdx.x;
  if (tid < OUT_DIM) acc3[tid] = 0.0f;
  __syncthreads();
  float a0 = 0.0f, a1 = 0.0f, a2 = 0.0f;
  const u32* hrow = (const u32*)(h + (size_t)node * HID);   // bf16 pairs
  for (int p = tid; p < HID / 2; p += 128) {                // p < 192
    const u32 pr = hrow[p];
    const float h0 = bf2f((u16)(pr & 0xFFFFu));
    const float h1 = bf2f((u16)(pr >> 16));
    const int k = 2 * p;
    a0 += h0 * Wp2[k]           + h1 * Wp2[k + 1];
    a1 += h0 * Wp2[HID + k]     + h1 * Wp2[HID + k + 1];
    a2 += h0 * Wp2[2 * HID + k] + h1 * Wp2[2 * HID + k + 1];
  }
  for (int m = 1; m < 32; m <<= 1) {
    a0 += __shfl_xor(a0, m);
    a1 += __shfl_xor(a1, m);
    a2 += __shfl_xor(a2, m);
  }
  if ((tid & 31) == 0) {
    atomicAdd(&acc3[0], a0);
    atomicAdd(&acc3[1], a1);
    atomicAdd(&acc3[2], a2);
  }
  __syncthreads();
  if (tid < OUT_DIM) out[(size_t)node * OUT_DIM + tid] = acc3[tid] + bp2[tid];
}

// ---------------------------------------------------------------------------
extern "C" void kernel_launch(void* const* d_in, const int* in_sizes, int n_in,
                              void* d_out, int out_size, void* d_ws, size_t ws_size,
                              hipStream_t stream) {
  (void)in_sizes; (void)n_in; (void)out_size; (void)ws_size;

  const float* x     = (const float*)d_in[0];
  const int*   ei    = (const int*)d_in[1];
  const float* W_l0  = (const float*)d_in[2];
  const float* b_l0  = (const float*)d_in[3];
  const float* W_r0  = (const float*)d_in[4];
  const float* W_l   = (const float*)d_in[5];
  const float* b_l   = (const float*)d_in[6];
  const float* W_r   = (const float*)d_in[7];
  const float* bn_g  = (const float*)d_in[8];
  const float* bn_b  = (const float*)d_in[9];
  const float* bn_m  = (const float*)d_in[10];
  const float* bn_v  = (const float*)d_in[11];
  const float* Wp0   = (const float*)d_in[12];
  const float* bp0   = (const float*)d_in[13];
  const float* Wp1   = (const float*)d_in[14];
  const float* bp1   = (const float*)d_in[15];
  const float* Wp2   = (const float*)d_in[16];
  const float* bp2   = (const float*)d_in[17];
  const float* pbn_g = (const float*)d_in[18];
  const float* pbn_b = (const float*)d_in[19];
  const float* pbn_m = (const float*)d_in[20];
  const float* pbn_v = (const float*)d_in[21];

  const int* src = ei;             // edge_index[0]
  const int* dst = ei + N_EDGES;   // edge_index[1]

  // workspace layout (all offsets 16B-aligned)
  u16*   h0  = (u16*)d_ws;                           // N*HID bf16 (76.8 MB)
  u16*   h1  = h0 + (size_t)N_NODES * HID;           // N*HID bf16
  float* agg = (float*)(h1 + (size_t)N_NODES * HID); // N*HID f32 (layer0: [N,10] f32)
  float* deg = agg + (size_t)N_NODES * HID;          // N f32
  u16*   pWl = (u16*)(deg + N_NODES);                // 147456 bf16
  u16*   pWr = pWl + (size_t)TILES * KSTEPS * 512;   // 147456 bf16

  // in-degrees (identical for every layer -> compute once)
  zero_kernel<<<256, 256, 0, stream>>>(deg, N_NODES);
  count_deg<<<(N_EDGES + 255) / 256, 256, 0, stream>>>(dst, deg);

  // ---- layer 0 (K = 10, VALU path) ----
  zero_kernel<<<256, 256, 0, stream>>>(agg, (long)N_NODES * IN_DIM);
  scatter10<<<(N_EDGES * IN_DIM + 255) / 256, 256, 0, stream>>>(src, dst, x, agg);
  layer0_kernel<<<N_NODES, 128, 0, stream>>>(x, agg, deg, W_l0, b_l0, W_r0,
                                             bn_g, bn_b, bn_m, bn_v, h0);

  // ---- layers 1..4 (WMMA path) ----
  u16* cur = h0;
  u16* nxt = h1;
  for (int t = 1; t < T_LAYERS; ++t) {
    zero_kernel<<<4096, 256, 0, stream>>>(agg, (long)N_NODES * HID);
    scatter384<<<dim3(N_EDGES, 3), 64, 0, stream>>>(src, dst, cur, agg);
    pack_weight<<<576, 256, 0, stream>>>(W_l + (size_t)(t - 1) * HID * HID, pWl);
    pack_weight<<<576, 256, 0, stream>>>(W_r + (size_t)(t - 1) * HID * HID, pWr);
    gemm_bn_act<true><<<N_NODES / 16, 256, 0, stream>>>(
        agg, cur, deg, pWl, pWr, b_l + (size_t)(t - 1) * HID,
        bn_g + (size_t)t * HID, bn_b + (size_t)t * HID,
        bn_m + (size_t)t * HID, bn_v + (size_t)t * HID, nxt);
    u16* tmp = cur; cur = nxt; nxt = tmp;
  }

  // ---- projection MLP (WMMA path, ReLU) ----
  pack_weight<<<576, 256, 0, stream>>>(Wp0, pWl);
  gemm_bn_act<false><<<N_NODES / 16, 256, 0, stream>>>(
      nullptr, cur, nullptr, pWl, nullptr, bp0, pbn_g, pbn_b, pbn_m, pbn_v, nxt);
  { u16* t2 = cur; cur = nxt; nxt = t2; }
  pack_weight<<<576, 256, 0, stream>>>(Wp1, pWl);
  gemm_bn_act<false><<<N_NODES / 16, 256, 0, stream>>>(
      nullptr, cur, nullptr, pWl, nullptr, bp1,
      pbn_g + HID, pbn_b + HID, pbn_m + HID, pbn_v + HID, nxt);
  { u16* t2 = cur; cur = nxt; nxt = t2; }

  proj3<<<N_NODES, 128, 0, stream>>>(cur, Wp2, bp2, (float*)d_out);
}
